// EdgeCompressedDGDN_9285719294448
// MI455X (gfx1250) — matrix operand order repaired and checked by reading, not compiled, not run on hardware
//
#include <hip/hip_runtime.h>
#include <hip/hip_bf16.h>
#include <stdint.h>

// ---------------------------------------------------------------------------
// EdgeCompressedDGDN for MI455X (gfx1250, wave32, WMMA f16 -> f32 accumulate)
//
//  * weights pre-packed once into WMMA B-fragment lane order (f16)
//  * hot kernels stage packed weights into LDS via async global->LDS copies
//    (ASYNCcnt path), B-fragments then come from ds_load_b128
//  * h kept in f32 for residual/LayerNorm + f16 shadow for all WMMA feeds
//  * full-tile fast paths: no per-element exec-mask branches in steady state
// ---------------------------------------------------------------------------

typedef _Float16 f16;
typedef __attribute__((ext_vector_type(16))) _Float16 v16h;
typedef __attribute__((ext_vector_type(8)))  _Float16 v8h;
typedef __attribute__((ext_vector_type(8)))  float    v8f;
typedef __attribute__((ext_vector_type(4)))  float    f32x4;
typedef __attribute__((ext_vector_type(4)))  int      v4i;

#define HID 128
#define OC_TILES 8      // 128 output cols / 16
#define NW 8            // waves per block (256 threads, wave32)

#define AS1 __attribute__((address_space(1)))
#define AS3 __attribute__((address_space(3)))

#if __has_builtin(__builtin_amdgcn_global_load_async_to_lds_b128)
#define HAS_ASYNC_CP 1
#else
#define HAS_ASYNC_CP 0
#endif

__device__ __forceinline__ int imin(int a, int b) { return a < b ? a : b; }
__device__ __forceinline__ int imax(int a, int b) { return a > b ? a : b; }

// 16B global -> LDS copy (per-lane addresses), async if available.
__device__ __forceinline__ void cp16_to_lds(const _Float16* g, _Float16* l) {
#if HAS_ASYNC_CP
  __builtin_amdgcn_global_load_async_to_lds_b128((AS1 v4i*)g, (AS3 v4i*)l, 0,
                                                 0);
#else
  *(f32x4*)l = *(const f32x4*)g;
#endif
}

__device__ __forceinline__ void wait_lds_cp() {
#if HAS_ASYNC_CP
#if __has_builtin(__builtin_amdgcn_s_wait_asynccnt)
  __builtin_amdgcn_s_wait_asynccnt(0);
#else
  asm volatile("s_wait_asynccnt 0x0" ::: "memory");
#endif
#endif
}

// Cooperative stage of `halfs` f16 elements into LDS (block-wide).
__device__ __forceinline__ void stage_w(const _Float16* __restrict__ g,
                                        _Float16* l, int halfs) {
  const int gran = halfs >> 3;  // 16B granules
  for (int i = threadIdx.x; i < gran; i += NW * 32)
    cp16_to_lds(g + (size_t)i * 8, l + (size_t)i * 8);
}

__device__ __forceinline__ v8f wmma_f16(v16h a, v16h b, v8f c) {
  return __builtin_amdgcn_wmma_f32_16x16x32_f16(false, a, false, b, (short)0, c,
                                                false, false);
}

__device__ __forceinline__ v8f bcast8(float x) {
  v8f c;
#pragma unroll
  for (int i = 0; i < 8; ++i) c[i] = x;
  return c;
}

// A-fragment (16x32 f16, ISA 7.12.2): lanes 0-15 K in {0..7,16..23},
// lanes 16-31 K in {8..15,24..31}.
__device__ __forceinline__ v16h load_a_f16(const _Float16* rowk, int lane) {
  const int base = (lane & 16) ? 8 : 0;
  v8h lo = *(const v8h*)(rowk + base);
  v8h hi = *(const v8h*)(rowk + base + 16);
  return __builtin_shufflevector(lo, hi, 0, 1, 2, 3, 4, 5, 6, 7, 8, 9, 10, 11,
                                 12, 13, 14, 15);
}

__device__ __forceinline__ v16h load_a_f32(const float* rowk, int lane) {
  const int base = (lane & 16) ? 8 : 0;
  v16h a;
#pragma unroll
  for (int i = 0; i < 8; ++i) {
    a[i]     = (_Float16)rowk[base + i];
    a[i + 8] = (_Float16)rowk[base + 16 + i];
  }
  return a;
}

// B-fragment: pre-packed lane-major, 32 lanes x 16 halfs per fragment.
// Works for both global and LDS-resident packs (generic pointer).
__device__ __forceinline__ v16h load_b_frag(const _Float16* P, int frag,
                                            int lane) {
  return *(const v16h*)(P + ((size_t)frag * 32 + lane) * 16);
}

// ---------------------------------------------------------------------------
// Pack W [O][K] (f32 row-major) into B-fragment order (f16): B = W^T.
// ---------------------------------------------------------------------------
__global__ void pack_w_kernel(const float* __restrict__ W,
                              _Float16* __restrict__ P, int K, int O) {
  int t = blockIdx.x * blockDim.x + threadIdx.x;
  if (t >= K * O) return;
  int e = t & 15;
  int L = (t >> 4) & 31;
  int frag = t >> 9;
  int OC = O >> 4;
  int oc = frag % OC;
  int kc = frag / OC;
  int kb = (L & 16) >> 1;
  int k = kc * 32 + kb + (e & 7) + ((e & 8) << 1);
  int o = oc * 16 + (L & 15);
  P[t] = (_Float16)W[(size_t)o * K + k];
}

__global__ void zero_f32_kernel(float* __restrict__ p, long n) {
  long i = (long)blockIdx.x * blockDim.x + threadIdx.x;
  if (i < n) p[i] = 0.0f;
}

// ---------------------------------------------------------------------------
// Encoder: h = x @ Wenc^T + b ; writes f32 h and f16 shadow hh.
// ---------------------------------------------------------------------------
__global__ __launch_bounds__(NW * 32) void encode_kernel(
    const float* __restrict__ x, const _Float16* __restrict__ Wp,
    const float* __restrict__ bias, float* __restrict__ h,
    _Float16* __restrict__ hh, int N) {
  const int lane = threadIdx.x & 31;
  const int wave = threadIdx.x >> 5;
  const int tile = blockIdx.x * NW + wave;
  const int ntiles = (N + 15) >> 4;
  if (tile >= ntiles) return;
  const int ln = lane & 15;
  const int hi = (lane >> 4) & 1;
  const bool full = (tile * 16 + 16 <= N);
  const int arow = imin(tile * 16 + ln, N - 1);
  const float* xr = x + (size_t)arow * HID;

  v16h a[4];
#pragma unroll
  for (int kc = 0; kc < 4; ++kc) a[kc] = load_a_f32(xr + kc * 32, lane);

#pragma unroll
  for (int oc = 0; oc < OC_TILES; ++oc) {
    v8f c = bcast8(bias[oc * 16 + ln]);
#pragma unroll
    for (int kc = 0; kc < 4; ++kc)
      c = wmma_f16(a[kc], load_b_frag(Wp, kc * OC_TILES + oc, lane), c);
    if (full) {
#pragma unroll
      for (int v = 0; v < 8; ++v) {
        size_t idx = (size_t)(tile * 16 + v + 8 * hi) * HID + oc * 16 + ln;
        h[idx] = c[v];
        hh[idx] = (_Float16)c[v];
      }
    } else {
#pragma unroll
      for (int v = 0; v < 8; ++v) {
        int node = tile * 16 + v + 8 * hi;
        if (node < N) {
          size_t idx = (size_t)node * HID + oc * 16 + ln;
          h[idx] = c[v];
          hh[idx] = (_Float16)c[v];
        }
      }
    }
  }
}

// ---------------------------------------------------------------------------
// Edge message kernel. One wave = 16 edges; weights staged in LDS per block.
//   m1 = relu(concat(h[row],h[col]) @ Wm1^T + b1)   (64 WMMA)
//   m2 = m1 @ Wm2^T + b2                            (32 WMMA)
//   agg[col] += m2  (global_atomic_add_f32 == segment_sum)
// ---------------------------------------------------------------------------
__global__ __launch_bounds__(NW * 32) void edge_msg_kernel(
    const int* __restrict__ row, const int* __restrict__ col,
    const _Float16* __restrict__ hh, const _Float16* __restrict__ W1p,
    const float* __restrict__ b1, const _Float16* __restrict__ W2p,
    const float* __restrict__ b2, float* __restrict__ agg, int E, int N) {
  __shared__ _Float16 w1s[256 * HID];   // 64 KB packed Wm1^T fragments
  __shared__ _Float16 w2s[HID * HID];   // 32 KB packed Wm2^T fragments
  __shared__ _Float16 lm[NW][16][HID];  // 32 KB relu(m1) tiles
  const int lane = threadIdx.x & 31;
  const int wave = threadIdx.x >> 5;
  const int tile = blockIdx.x * NW + wave;
  const int etiles = (E + 15) >> 4;
  const bool active = tile < etiles;
  const int tb = active ? tile * 16 : 0;
  const bool full = active && (tb + 16 <= E);
  const int ln = lane & 15;
  const int hi = (lane >> 4) & 1;

  // Issue async weight staging first; overlap with feature gathers below.
  stage_w(W1p, w1s, 256 * HID);
  stage_w(W2p, w2s, HID * HID);

  const int eL = imin(tb + ln, E - 1);
  const int rn = imin(imax(row[eL], 0), N - 1);
  const int cn = imin(imax(col[eL], 0), N - 1);  // lane r holds col[tb+r]
  const _Float16* hr = hh + (size_t)rn * HID;
  const _Float16* hc = hh + (size_t)cn * HID;

  v16h a[8];
#pragma unroll
  for (int kc = 0; kc < 4; ++kc) a[kc]     = load_a_f16(hr + kc * 32, lane);
#pragma unroll
  for (int kc = 0; kc < 4; ++kc) a[4 + kc] = load_a_f16(hc + kc * 32, lane);

  // Scatter destinations for this lane's 8 output rows via intra-wave shuffle.
  int dst[8];
#pragma unroll
  for (int v = 0; v < 8; ++v) dst[v] = __shfl(cn, v + 8 * hi, 32);

  wait_lds_cp();
  __syncthreads();  // weights resident in LDS

#pragma unroll
  for (int oc = 0; oc < OC_TILES; ++oc) {
    v8f c = bcast8(b1[oc * 16 + ln]);
#pragma unroll
    for (int kc = 0; kc < 8; ++kc)
      c = wmma_f16(a[kc], load_b_frag(w1s, kc * OC_TILES + oc, lane), c);
#pragma unroll
    for (int v = 0; v < 8; ++v) {
      float z = c[v] > 0.0f ? c[v] : 0.0f;  // ReLU
      lm[wave][v + 8 * hi][oc * 16 + ln] = (_Float16)z;
    }
  }
  __syncthreads();  // lm tiles complete

  v16h a2[4];
#pragma unroll
  for (int kc = 0; kc < 4; ++kc)
    a2[kc] = load_a_f16(&lm[wave][ln][kc * 32], lane);

#pragma unroll
  for (int oc = 0; oc < OC_TILES; ++oc) {
    v8f c = bcast8(b2[oc * 16 + ln]);
#pragma unroll
    for (int kc = 0; kc < 4; ++kc)
      c = wmma_f16(a2[kc], load_b_frag(w2s, kc * OC_TILES + oc, lane), c);
    if (full) {
#pragma unroll
      for (int v = 0; v < 8; ++v)
        __hip_atomic_fetch_add(&agg[(size_t)dst[v] * HID + oc * 16 + ln], c[v],
                               __ATOMIC_RELAXED, __HIP_MEMORY_SCOPE_AGENT);
    } else {
#pragma unroll
      for (int v = 0; v < 8; ++v) {
        if (active && (tb + v + 8 * hi) < E)
          __hip_atomic_fetch_add(&agg[(size_t)dst[v] * HID + oc * 16 + ln],
                                 c[v], __ATOMIC_RELAXED,
                                 __HIP_MEMORY_SCOPE_AGENT);
      }
    }
  }
}

// ---------------------------------------------------------------------------
// Node update: upd = concat(h, agg) @ Wu^T + bu ; h = LN(h + upd).
// Wu staged in LDS; residual tile staged in LDS (f32); lanes 0-15 LN one row.
// ---------------------------------------------------------------------------
__global__ __launch_bounds__(NW * 32) void node_update_kernel(
    const _Float16* __restrict__ hh, const float* __restrict__ agg,
    const _Float16* __restrict__ Wup, const float* __restrict__ bu,
    const float* __restrict__ gamma, const float* __restrict__ beta,
    float* __restrict__ h, _Float16* __restrict__ hhout, int N) {
  __shared__ _Float16 wus[256 * HID];  // 64 KB packed Wu^T fragments
  __shared__ float lt[NW][16][HID];    // 64 KB residual tiles
  const int lane = threadIdx.x & 31;
  const int wave = threadIdx.x >> 5;
  const int tile = blockIdx.x * NW + wave;
  const int ntiles = (N + 15) >> 4;
  const bool active = tile < ntiles;
  const int tb = active ? tile * 16 : 0;
  const bool full = active && (tb + 16 <= N);
  const int ln = lane & 15;
  const int hi = (lane >> 4) & 1;
  const int arow = imin(tb + ln, N - 1);

  stage_w(Wup, wus, 256 * HID);

  v16h a[8];
#pragma unroll
  for (int kc = 0; kc < 4; ++kc)
    a[kc] = load_a_f16(hh + (size_t)arow * HID + kc * 32, lane);
#pragma unroll
  for (int kc = 0; kc < 4; ++kc)
    a[4 + kc] = load_a_f32(agg + (size_t)arow * HID + kc * 32, lane);

  wait_lds_cp();
  __syncthreads();  // weights resident in LDS

#pragma unroll
  for (int oc = 0; oc < OC_TILES; ++oc) {
    v8f c = bcast8(bu[oc * 16 + ln]);
#pragma unroll
    for (int kc = 0; kc < 8; ++kc)
      c = wmma_f16(a[kc], load_b_frag(wus, kc * OC_TILES + oc, lane), c);
    if (full) {
#pragma unroll
      for (int v = 0; v < 8; ++v) {
        int node = tb + v + 8 * hi;
        lt[wave][v + 8 * hi][oc * 16 + ln] =
            c[v] + h[(size_t)node * HID + oc * 16 + ln];
      }
    } else {
#pragma unroll
      for (int v = 0; v < 8; ++v) {
        int node = imin(tb + v + 8 * hi, N - 1);
        lt[wave][v + 8 * hi][oc * 16 + ln] =
            c[v] + h[(size_t)node * HID + oc * 16 + ln];
      }
    }
  }
  __syncthreads();

  if (lane < 16) {
    int node = tb + lane;
    if (active && node < N) {
      float s = 0.0f, s2 = 0.0f;
#pragma unroll 4
      for (int k = 0; k < HID; ++k) {
        float v = lt[wave][lane][k];
        s += v;
        s2 += v * v;
      }
      float mu = s * (1.0f / HID);
      float var = s2 * (1.0f / HID) - mu * mu;
      float inv = rsqrtf(var + 1e-5f);
#pragma unroll 4
      for (int k = 0; k < HID; ++k) {
        float v = (lt[wave][lane][k] - mu) * inv * gamma[k] + beta[k];
        size_t idx = (size_t)node * HID + k;
        h[idx] = v;
        hhout[idx] = (_Float16)v;
      }
    }
  }
}

// ---------------------------------------------------------------------------
// Output projection: out = h @ Wout^T + b  (A from f16 shadow, out f32)
// ---------------------------------------------------------------------------
__global__ __launch_bounds__(NW * 32) void output_kernel(
    const _Float16* __restrict__ hh, const _Float16* __restrict__ Wp,
    const float* __restrict__ bias, float* __restrict__ out, int N) {
  const int lane = threadIdx.x & 31;
  const int wave = threadIdx.x >> 5;
  const int tile = blockIdx.x * NW + wave;
  const int ntiles = (N + 15) >> 4;
  if (tile >= ntiles) return;
  const int ln = lane & 15;
  const int hi = (lane >> 4) & 1;
  const bool full = (tile * 16 + 16 <= N);
  const int arow = imin(tile * 16 + ln, N - 1);

  v16h a[4];
#pragma unroll
  for (int kc = 0; kc < 4; ++kc)
    a[kc] = load_a_f16(hh + (size_t)arow * HID + kc * 32, lane);

#pragma unroll
  for (int oc = 0; oc < OC_TILES; ++oc) {
    v8f c = bcast8(bias[oc * 16 + ln]);
#pragma unroll
    for (int kc = 0; kc < 4; ++kc)
      c = wmma_f16(a[kc], load_b_frag(Wp, kc * OC_TILES + oc, lane), c);
    if (full) {
#pragma unroll
      for (int v = 0; v < 8; ++v)
        out[(size_t)(tile * 16 + v + 8 * hi) * HID + oc * 16 + ln] = c[v];
    } else {
#pragma unroll
      for (int v = 0; v < 8; ++v) {
        int node = tile * 16 + v + 8 * hi;
        if (node < N) out[(size_t)node * HID + oc * 16 + ln] = c[v];
      }
    }
  }
}

// ---------------------------------------------------------------------------
extern "C" void kernel_launch(void* const* d_in, const int* in_sizes, int n_in,
                              void* d_out, int out_size, void* d_ws,
                              size_t ws_size, hipStream_t stream) {
  const float* x     = (const float*)d_in[0];
  const int*   ei    = (const int*)d_in[1];
  const float* Wenc  = (const float*)d_in[2];
  const float* benc  = (const float*)d_in[3];
  const float* Wm1   = (const float*)d_in[4];
  const float* bm1   = (const float*)d_in[5];
  const float* Wm2   = (const float*)d_in[6];
  const float* bm2   = (const float*)d_in[7];
  const float* Wu    = (const float*)d_in[8];
  const float* bu    = (const float*)d_in[9];
  const float* gamma = (const float*)d_in[10];
  const float* beta  = (const float*)d_in[11];
  const float* Wout  = (const float*)d_in[12];
  const float* bout  = (const float*)d_in[13];

  const int N = in_sizes[0] / HID;
  const int E = in_sizes[1] / 2;
  const int* row = ei;
  const int* col = ei + E;

  char* base = (char*)d_ws;
  size_t off = 0;
  auto alloc = [&](size_t bytes) -> void* {
    void* p = base + off;
    off = (off + bytes + 255) & ~(size_t)255;
    return p;
  };
  float*    h     = (float*)alloc((size_t)N * HID * sizeof(float));
  float*    agg   = (float*)alloc((size_t)N * HID * sizeof(float));
  _Float16* hh    = (_Float16*)alloc((size_t)N * HID * sizeof(_Float16));
  _Float16* WencP = (_Float16*)alloc((size_t)128 * 128 * 2);
  _Float16* Wm1P0 = (_Float16*)alloc((size_t)256 * 128 * 2);
  _Float16* Wm1P1 = (_Float16*)alloc((size_t)256 * 128 * 2);
  _Float16* Wm2P0 = (_Float16*)alloc((size_t)128 * 128 * 2);
  _Float16* Wm2P1 = (_Float16*)alloc((size_t)128 * 128 * 2);
  _Float16* WuP0  = (_Float16*)alloc((size_t)256 * 128 * 2);
  _Float16* WuP1  = (_Float16*)alloc((size_t)256 * 128 * 2);
  _Float16* WoutP = (_Float16*)alloc((size_t)128 * 128 * 2);
  (void)ws_size; (void)n_in; (void)out_size;

  _Float16* Wm1P[2] = {Wm1P0, Wm1P1};
  _Float16* Wm2P[2] = {Wm2P0, Wm2P1};
  _Float16* WuP[2]  = {WuP0, WuP1};

  auto packKO = [&](const float* W, _Float16* P, int K, int O) {
    int total = K * O;
    pack_w_kernel<<<(total + 255) / 256, 256, 0, stream>>>(W, P, K, O);
  };
  packKO(Wenc, WencP, 128, 128);
  for (int l = 0; l < 2; ++l) {
    packKO(Wm1 + (size_t)l * 128 * 256, Wm1P[l], 256, 128);
    packKO(Wm2 + (size_t)l * 128 * 128, Wm2P[l], 128, 128);
    packKO(Wu  + (size_t)l * 128 * 256, WuP[l],  256, 128);
  }
  packKO(Wout, WoutP, 128, 128);

  const int ntiles = (N + 15) / 16;
  const int etiles = (E + 15) / 16;
  const int nblocks = (ntiles + NW - 1) / NW;
  const int eblocks = (etiles + NW - 1) / NW;
  const dim3 blk(NW * 32);

  encode_kernel<<<nblocks, blk, 0, stream>>>(x, WencP, benc, h, hh, N);

  for (int l = 0; l < 2; ++l) {
    long nAgg = (long)N * HID;
    zero_f32_kernel<<<(int)((nAgg + 255) / 256), 256, 0, stream>>>(agg, nAgg);
    edge_msg_kernel<<<eblocks, blk, 0, stream>>>(
        row, col, hh, Wm1P[l], bm1 + (size_t)l * HID, Wm2P[l],
        bm2 + (size_t)l * HID, agg, E, N);
    node_update_kernel<<<nblocks, blk, 0, stream>>>(
        hh, agg, WuP[l], bu + (size_t)l * HID, gamma + (size_t)l * HID,
        beta + (size_t)l * HID, h, hh, N);
  }

  output_kernel<<<nblocks, blk, 0, stream>>>(hh, WoutP, bout, (float*)d_out, N);
}
